// TFNInteractionBlock_47571057770948
// MI455X (gfx1250) — matrix operand order repaired and checked by reading, not compiled, not run on hardware
//
#include <hip/hip_runtime.h>
#include <hip/hip_bf16.h>
#include <math.h>

#define N_NODES   8000
#define N_EDGES   64000
#define RAD_HID   64
#define WNUMEL    2304

// sqrt(1/48), sqrt(3/48)=0.25, 1/sqrt(3)
#define PW0_C   0.14433756729740643f
#define PW1_C   0.25f
#define ISQ3_C  0.57735026918962576f
#define C_VB    (PW1_C * ISQ3_C)      /* = 0.14433757 */
#define C_WD    (PW0_C * ISQ3_C)      /* = 1/12       */

typedef __attribute__((ext_vector_type(16))) _Float16 v16h;
typedef __attribute__((ext_vector_type(8)))  float    v8f;

union Frag16 { v16h h; unsigned int u[8]; };

__device__ __forceinline__ float silu_f(float x) { return x / (1.f + __expf(-x)); }

// ---------------- kernel 0: zero the aggregation buffer --------------------
__global__ void zero_kernel(float* __restrict__ p, int n) {
    int i = blockIdx.x * blockDim.x + threadIdx.x;
    if (i < n) p[i] = 0.f;
}

// ---------------- kernel 1: radial layer 1, h = silu(emb@W1+b1) as f16 -----
__global__ void radial1_kernel(const float* __restrict__ emb,
                               const float* __restrict__ W1,
                               const float* __restrict__ b1,
                               _Float16* __restrict__ h) {
    int i = blockIdx.x * blockDim.x + threadIdx.x;          // E*64
    if (i >= N_EDGES * RAD_HID) return;
    int e = i >> 6, j = i & 63;
    const float* er = emb + e * 8;
    float acc = b1[j];
#pragma unroll
    for (int b = 0; b < 8; ++b) acc += er[b] * W1[b * RAD_HID + j];
    h[i] = (_Float16)silu_f(acc);
}

// ---------------- kernel 2: transpose rW2 (64x2304) -> f16 (2304x64) -------
__global__ void tw2_kernel(const float* __restrict__ w2, _Float16* __restrict__ w2t) {
    int i = blockIdx.x * blockDim.x + threadIdx.x;          // 2304*64
    if (i >= WNUMEL * RAD_HID) return;
    int n = i >> 6, k = i & 63;
    w2t[i] = (_Float16)w2[k * WNUMEL + n];
}

// ---------------- kernel 3: fused GEMM + message build + scatter -----------
// grid = 125 blocks x 128 threads (4 waves). 500 waves x 8 tiles = 4000 tiles.
__global__ void __launch_bounds__(128)
edge_fused_kernel(const float* __restrict__ nf,
                  const long long* __restrict__ ei,
                  const float* __restrict__ esh,
                  const float* __restrict__ rb2,
                  const _Float16* __restrict__ h,
                  const _Float16* __restrict__ w2t,
                  float* __restrict__ agg) {
    extern __shared__ char smem[];
    const int tid  = threadIdx.x;
    const int wave = tid >> 5;
    const int lane = tid & 31;
    const int col  = lane & 15;
    const int hi   = lane >> 4;

    // ---- cooperative load of transposed rW2 (f16) into LDS (288 KB) ----
    {
        const uint4* g = (const uint4*)w2t;
        uint4*       l = (uint4*)smem;
        const int n16 = (WNUMEL * RAD_HID * 2) / 16;        // 18432
        for (int i = tid; i < n16; i += 128) l[i] = g[i];
    }

    // per-wave scratch: 1664 floats each
    float* wbase = (float*)(smem + WNUMEL * RAD_HID * 2) + wave * 1664;
    float* axs  = wbase;            // [16][32] xs per edge
    float* xvb  = wbase + 512;      // [16][16][3]
    float* dvb  = wbase + 1280;     // [16][16] dotv
    float* s0b  = wbase + 1536;     // [16] sh0
    float* svb  = wbase + 1552;     // [16][3] shv
    int*   dstb = (int*)(wbase + 1600);

    const unsigned int* wl = (const unsigned int*)smem;     // rW2T as dwords
    const int gwave = blockIdx.x * 4 + wave;                // 0..499

#pragma unroll 1
    for (int it = 0; it < 8; ++it) {
        const int e0 = (gwave + it * 500) * 16;

        __syncthreads();                                    // protect LDS reuse
        if (lane < 16) {                                    // one edge per lane
            const int e = e0 + lane;
            const long long s = ei[e];
            dstb[lane] = (int)ei[N_EDGES + e];
            const float* row = nf + s * 80;
            float* ax = axs + lane * 32;
#pragma unroll
            for (int u = 0; u < 32; ++u) ax[u] = row[u];
            const float sh0 = esh[e * 4 + 0];
            const float v0 = esh[e * 4 + 1], v1 = esh[e * 4 + 2], v2 = esh[e * 4 + 3];
            s0b[lane] = sh0;
            svb[lane * 3 + 0] = v0; svb[lane * 3 + 1] = v1; svb[lane * 3 + 2] = v2;
            float* xv = xvb + lane * 48;
            float* dv = dvb + lane * 16;
#pragma unroll
            for (int u = 0; u < 16; ++u) {
                float x0 = row[32 + u * 3 + 0];
                float x1 = row[32 + u * 3 + 1];
                float x2 = row[32 + u * 3 + 2];
                xv[u * 3 + 0] = x0; xv[u * 3 + 1] = x1; xv[u * 3 + 2] = x2;
                dv[u] = x0 * v0 + x1 * v1 + x2 * v2;
            }
        }
        __syncthreads();

        // ---- A fragments: h tile, 16 edges x 64 K (two 16x32 f16 frags) ----
        Frag16 a0, a1;
        {
            const _Float16* hrow = h + (long)(e0 + col) * 64;
#pragma unroll
            for (int r = 0; r < 8; ++r) {
                int k0 = (hi << 4) + (r << 1);
                a0.u[r] = *(const unsigned int*)(hrow + k0);
                a1.u[r] = *(const unsigned int*)(hrow + 32 + k0);
            }
        }

        // per-lane cached edge scalars for its 8 rows (edge = r + 8*hi)
        float sh0c[8], sv0c[8], sv1c[8], sv2c[8];
#pragma unroll
        for (int r = 0; r < 8; ++r) {
            int e_r = r + (hi << 3);
            sh0c[r] = s0b[e_r];
            sv0c[r] = svb[e_r * 3 + 0];
            sv1c[r] = svb[e_r * 3 + 1];
            sv2c[r] = svb[e_r * 3 + 2];
        }

        float ms0[8] = {0}, ms1[8] = {0};
        float mv0[8] = {0}, mv1[8] = {0}, mv2[8] = {0};

        // one 16(edge)x16(col) tile of tpw, with bias folded into C
        auto tpwTile = [&](int t) -> v8f {
            float bias = rb2[t * 16 + col];
            v8f acc = {bias, bias, bias, bias, bias, bias, bias, bias};
            Frag16 b0, b1;
            const unsigned int* wr = wl + (t * 16 + col) * 32 + (hi << 3);
#pragma unroll
            for (int r = 0; r < 8; ++r) { b0.u[r] = wr[r]; b1.u[r] = wr[16 + r]; }
            acc = __builtin_amdgcn_wmma_f32_16x16x32_f16(false, a0.h, false, b0.h,
                                                         (short)0, acc, false, false);
            acc = __builtin_amdgcn_wmma_f32_16x16x32_f16(false, a1.h, false, b1.h,
                                                         (short)0, acc, false, false);
            return acc;
        };

        // ---- wA region: t = 0..63 : u = t>>1, w-half = t&1 ----
        for (int tt = 0; tt < 32; ++tt) {
            {   v8f d = tpwTile(2 * tt);
#pragma unroll
                for (int r = 0; r < 8; ++r) {
                    int e_r = r + (hi << 3);
                    ms0[r] += PW0_C * axs[e_r * 32 + tt] * sh0c[r] * d[r];
                } }
            {   v8f d = tpwTile(2 * tt + 1);
#pragma unroll
                for (int r = 0; r < 8; ++r) {
                    int e_r = r + (hi << 3);
                    ms1[r] += PW0_C * axs[e_r * 32 + tt] * sh0c[r] * d[r];
                } }
        }
        // ---- wB region: t = 64..95 : u = t-64, outer-product with shv ----
        for (int u = 0; u < 32; ++u) {
            v8f d = tpwTile(64 + u);
#pragma unroll
            for (int r = 0; r < 8; ++r) {
                int e_r = r + (hi << 3);
                float p = C_VB * axs[e_r * 32 + u] * d[r];
                mv0[r] += p * sv0c[r];
                mv1[r] += p * sv1c[r];
                mv2[r] += p * sv2c[r];
            }
        }
        // ---- wC region: t = 96..111 : u = t-96, times sh0 * xv ----
        for (int u = 0; u < 16; ++u) {
            v8f d = tpwTile(96 + u);
#pragma unroll
            for (int r = 0; r < 8; ++r) {
                int e_r = r + (hi << 3);
                float q = C_VB * sh0c[r] * d[r];
                const float* xv = xvb + e_r * 48 + u * 3;
                mv0[r] += q * xv[0];
                mv1[r] += q * xv[1];
                mv2[r] += q * xv[2];
            }
        }
        // ---- wD region: t = 112..143 : u = (t-112)>>1, w-half = (t-112)&1 ----
        for (int tt = 0; tt < 16; ++tt) {
            {   v8f d = tpwTile(112 + 2 * tt);
#pragma unroll
                for (int r = 0; r < 8; ++r) {
                    int e_r = r + (hi << 3);
                    ms0[r] += C_WD * dvb[e_r * 16 + tt] * d[r];
                } }
            {   v8f d = tpwTile(112 + 2 * tt + 1);
#pragma unroll
                for (int r = 0; r < 8; ++r) {
                    int e_r = r + (hi << 3);
                    ms1[r] += C_WD * dvb[e_r * 16 + tt] * d[r];
                } }
        }

        // ---- scatter-add into agg[dst][80] ----
#pragma unroll
        for (int r = 0; r < 8; ++r) {
            int e_r = r + (hi << 3);
            float* row = agg + (long)dstb[e_r] * 80;
            atomicAdd(row + col,        ms0[r]);
            atomicAdd(row + 16 + col,   ms1[r]);
            float* vp = row + 32 + col * 3;
            atomicAdd(vp + 0, mv0[r]);
            atomicAdd(vp + 1, mv1[r]);
            atomicAdd(vp + 2, mv2[r]);
        }
    }
}

// ---------------- kernel 4: node update (linear + gated nonlinearity) ------
__global__ void node_update_kernel(const float* __restrict__ agg,
                                   const float* __restrict__ nf,
                                   const float* __restrict__ Wl0,
                                   const float* __restrict__ Wl1,
                                   float* __restrict__ out) {
    int i = blockIdx.x * blockDim.x + threadIdx.x;          // 8000*48
    if (i >= N_NODES * 48) return;
    int n = i / 48;
    int q = i - n * 48;
    const float* arow = agg + n * 80;
    if (q < 32) {
        float acc = 0.f;
#pragma unroll
        for (int u = 0; u < 32; ++u) acc += arow[u] * Wl0[u * 32 + q];
        acc *= 0.17677669529663689f;                        // 1/sqrt(32)
        out[n * 80 + q] = nf[n * 80 + q] + silu_f(acc);
    } else {
        int v = q - 32;
        float l0 = 0.f, l1 = 0.f, l2 = 0.f;
#pragma unroll
        for (int u = 0; u < 16; ++u) {
            float w = Wl1[u * 16 + v];
            l0 += arow[32 + u * 3 + 0] * w;
            l1 += arow[32 + u * 3 + 1] * w;
            l2 += arow[32 + u * 3 + 2] * w;
        }
        l0 *= 0.25f; l1 *= 0.25f; l2 *= 0.25f;              // 1/sqrt(16)
        float nrm = sqrtf(l0 * l0 + l1 * l1 + l2 * l2);
        float o0 = 0.f, o1 = 0.f, o2 = 0.f;
        if (nrm >= 1e-8f) {
            float g = silu_f(nrm) / nrm;                    // safe == nrm here
            o0 = l0 * g; o1 = l1 * g; o2 = l2 * g;
        }
        int base = n * 80 + 32 + v * 3;
        out[base + 0] = nf[base + 0] + o0;
        out[base + 1] = nf[base + 1] + o1;
        out[base + 2] = nf[base + 2] + o2;
    }
}

// ---------------------------------------------------------------------------
extern "C" void kernel_launch(void* const* d_in, const int* in_sizes, int n_in,
                              void* d_out, int out_size, void* d_ws, size_t ws_size,
                              hipStream_t stream) {
    const float*     nf   = (const float*)d_in[0];
    const long long* ei   = (const long long*)d_in[1];
    const float*     esh  = (const float*)d_in[2];
    const float*     emb  = (const float*)d_in[3];
    const float*     rW1  = (const float*)d_in[4];
    const float*     rb1  = (const float*)d_in[5];
    const float*     rW2  = (const float*)d_in[6];
    const float*     rb2  = (const float*)d_in[7];
    const float*     Wl0  = (const float*)d_in[8];
    const float*     Wl1  = (const float*)d_in[9];
    float* out = (float*)d_out;

    char* ws = (char*)d_ws;
    float*     agg = (float*)ws;                                   // 8000*80*4   = 2,560,000 B
    _Float16*  hbf = (_Float16*)(ws + 2560000);                    // 64000*64*2  = 8,192,000 B
    _Float16*  w2t = (_Float16*)(ws + 2560000 + 8192000);          // 2304*64*2   =   294,912 B

    zero_kernel<<<(N_NODES * 80 + 255) / 256, 256, 0, stream>>>(agg, N_NODES * 80);
    radial1_kernel<<<(N_EDGES * RAD_HID + 255) / 256, 256, 0, stream>>>(emb, rW1, rb1, hbf);
    tw2_kernel<<<(WNUMEL * RAD_HID + 255) / 256, 256, 0, stream>>>(rW2, w2t);

    size_t lds = (size_t)WNUMEL * RAD_HID * 2 + 4 * 1664 * 4;      // 294,912 + 26,624 = 321,536 B
    edge_fused_kernel<<<125, 128, lds, stream>>>(nf, ei, esh, rb2, hbf, w2t, agg);

    node_update_kernel<<<(N_NODES * 48 + 255) / 256, 256, 0, stream>>>(agg, nf, Wl0, Wl1, out);
}
